// Model_21122649162464
// MI455X (gfx1250) — matrix operand order repaired and checked by reading, not compiled, not run on hardware
//
#include <hip/hip_runtime.h>
#include <hip/hip_bf16.h>
#include <math.h>

typedef __bf16 bf16_t;
typedef __attribute__((ext_vector_type(16))) __bf16 v16bf;
typedef __attribute__((ext_vector_type(8)))  __bf16 v8bf;
typedef __attribute__((ext_vector_type(8)))  float  v8f;

constexpr int B = 32, S = 512, T = 64, E = 256, H = 512, V = 30000;
constexpr int TH3 = 3 * H;      // 1536
constexpr int XK  = E + H;      // 768  decoder GRU input width
constexpr int FK  = E + 2 * H;  // 1280 logits feature width
constexpr int TD  = T - 1;      // 63 decode steps
constexpr int MF  = B * TD;     // 2016 valid feature rows
constexpr int MFP = 2048;       // padded to multiple of 512 (no row guards in GEMM)
constexpr int NWG_ENC = 8;
constexpr int NWG_DEC = 32;

// ---------------- workspace layout (bytes; every chunk 256B-aligned) --------
constexpr size_t OFF_BAR    = 0;                                   // 8 ints
constexpr size_t OFF_H0     = 256;                                 // 2*B*H f32 ping-pong
constexpr size_t OFF_H1     = OFF_H0     + (size_t)2*B*H*4;
constexpr size_t OFF_DH     = OFF_H1     + (size_t)2*B*H*4;
constexpr size_t OFF_H0B    = OFF_DH     + (size_t)2*B*H*4;        // bf16 shadows
constexpr size_t OFF_H1B    = OFF_H0B    + (size_t)2*B*H*2;
constexpr size_t OFF_DHB    = OFF_H1B    + (size_t)2*B*H*2;
constexpr size_t OFF_Q      = OFF_DHB    + (size_t)2*B*H*2;        // B*H f32
constexpr size_t OFF_SC     = OFF_Q      + (size_t)B*H*4;          // B*S f32
constexpr size_t OFF_XCAT   = OFF_SC     + (size_t)B*S*4;          // B*XK bf16
constexpr size_t OFF_SRCEMB = OFF_XCAT   + (size_t)B*XK*2;         // B*S*E bf16
constexpr size_t OFF_DECEMB = OFF_SRCEMB + (size_t)B*S*E*2;        // B*TD*E bf16
constexpr size_t OFF_GI0    = OFF_DECEMB + (size_t)B*TD*E*2;       // B*S*3H f32
constexpr size_t OFF_ENCRAW = OFF_GI0    + (size_t)B*S*TH3*4;      // B*S*H bf16
constexpr size_t OFF_ENCOUT = OFF_ENCRAW + (size_t)B*S*H*2;        // B*S*H f32
constexpr size_t OFF_ENCOUTB= OFF_ENCOUT + (size_t)B*S*H*4;        // B*S*H bf16
constexpr size_t OFF_ENCPRJ = OFF_ENCOUTB+ (size_t)B*S*H*2;        // B*S*H f32
constexpr size_t OFF_FEATS  = OFF_ENCPRJ + (size_t)B*S*H*4;        // MFP*FK bf16
constexpr size_t OFF_WIH0   = OFF_FEATS  + (size_t)MFP*FK*2;
constexpr size_t OFF_WHH0   = OFF_WIH0   + (size_t)TH3*E*2;
constexpr size_t OFF_WIH1   = OFF_WHH0   + (size_t)TH3*H*2;
constexpr size_t OFF_WHH1   = OFF_WIH1   + (size_t)TH3*H*2;
constexpr size_t OFF_LINW   = OFF_WHH1   + (size_t)TH3*H*2;
constexpr size_t OFF_ATTNW  = OFF_LINW   + (size_t)H*H*2;
constexpr size_t OFF_DWIH   = OFF_ATTNW  + (size_t)H*XK*2;
constexpr size_t OFF_DWHH   = OFF_DWIH   + (size_t)TH3*XK*2;
constexpr size_t OFF_OUTW   = OFF_DWHH   + (size_t)TH3*H*2;        // V*FK bf16

// ---------------- WMMA helpers ---------------------------------------------
__device__ inline v8f wmma_bf(v16bf a, v16bf b, v8f c) {
  return __builtin_amdgcn_wmma_f32_16x16x32_bf16(false, a, false, b, (short)0, c,
                                                 false, false);
}

// A fragment 16x32 bf16, row-major src: lane l<16 -> row l, K {k0..k0+7, k0+16..23};
// lanes 16..31 -> K {k0+8..15, k0+24..31}.
__device__ inline v16bf load_a_bf(const bf16_t* A, int lda, int row0, int k0) {
  int l = threadIdx.x & 31;
  const bf16_t* p = A + (size_t)(row0 + (l & 15)) * lda + k0 + ((l >> 4) ? 8 : 0);
  v8bf lo = *(const v8bf*)p;
  v8bf hi = *(const v8bf*)(p + 16);
  v16bf r;
#pragma unroll
  for (int i = 0; i < 8; ++i) { r[i] = lo[i]; r[i + 8] = hi[i]; }
  return r;
}

// B fragment 32x16 from weight W (N x K row-major): B[k][n] = W[n][k].
// lane l<16 -> col l, K {k0..k0+7, k0+16..23}; lanes 16..31 -> K {k0+8..15, k0+24..31}.
__device__ inline v16bf load_b_bf(const bf16_t* W, int ldb, int n0, int k0) {
  int l = threadIdx.x & 31;
  const bf16_t* p = W + (size_t)(n0 + (l & 15)) * ldb + k0 + ((l >> 4) ? 8 : 0);
  v8bf lo = *(const v8bf*)p;
  v8bf hi = *(const v8bf*)(p + 16);
  v16bf r;
#pragma unroll
  for (int i = 0; i < 8; ++i) { r[i] = lo[i]; r[i + 8] = hi[i]; }
  return r;
}

__device__ inline float gru_gate(float gr, float gz, float gn,
                                 float hr, float hz, float hn, float hold) {
  float r = 1.f / (1.f + __expf(-(gr + hr)));
  float z = 1.f / (1.f + __expf(-(gz + hz)));
  float n = tanhf(gn + r * hn);
  return (1.f - z) * n + z * hold;
}

// device-scope sense-reversing barrier across persistent workgroups
__device__ inline void gbarrier(int* cnt, int* gen, int nwg) {
  __syncthreads();
  if (threadIdx.x == 0) {
    __builtin_amdgcn_fence(__ATOMIC_RELEASE, "agent");
    int g = __hip_atomic_load(gen, __ATOMIC_RELAXED, __HIP_MEMORY_SCOPE_AGENT);
    int v = __hip_atomic_fetch_add(cnt, 1, __ATOMIC_RELAXED, __HIP_MEMORY_SCOPE_AGENT);
    if (v == nwg - 1) {
      __hip_atomic_store(cnt, 0, __ATOMIC_RELAXED, __HIP_MEMORY_SCOPE_AGENT);
      __hip_atomic_fetch_add(gen, 1, __ATOMIC_RELEASE, __HIP_MEMORY_SCOPE_AGENT);
    } else {
      while (__hip_atomic_load(gen, __ATOMIC_RELAXED, __HIP_MEMORY_SCOPE_AGENT) == g)
        __builtin_amdgcn_s_sleep(8);
    }
    __builtin_amdgcn_fence(__ATOMIC_ACQUIRE, "agent");
  }
  __syncthreads();
}

// ---------------- small utility kernels -------------------------------------
__global__ void k_init(char* ws) {
  int i = blockIdx.x * blockDim.x + threadIdx.x;
  if (i < 8) ((int*)(ws + OFF_BAR))[i] = 0;
  if (i < B * H) {
    ((float*)(ws + OFF_H0))[i] = 0.f;   // encoder h0 parity-0
    ((float*)(ws + OFF_H1))[i] = 0.f;   // encoder h1 parity-0
    ((bf16_t*)(ws + OFF_H0B))[i] = (bf16_t)0.f;
    ((bf16_t*)(ws + OFF_H1B))[i] = (bf16_t)0.f;
  }
  if (i < (MFP - MF) * FK)              // zero padded feats rows
    ((bf16_t*)(ws + OFF_FEATS))[(size_t)MF * FK + i] = (bf16_t)0.f;
}

__global__ void k_f2bf(const float* s, bf16_t* d, int n) {
  int i = blockIdx.x * blockDim.x + threadIdx.x;
  if (i < n) d[i] = (bf16_t)s[i];
}

__global__ void k_gather_src(const int* src, const float* emb, bf16_t* dst) {
  int i = blockIdx.x * blockDim.x + threadIdx.x;
  if (i >= B * S * E) return;
  int row = i >> 8, e = i & (E - 1);
  int tok = src[row];
  dst[i] = (bf16_t)emb[(size_t)tok * E + e];
}

__global__ void k_gather_tgt(const int* tgt, const float* emb, bf16_t* dembs,
                             bf16_t* feats) {
  int i = blockIdx.x * blockDim.x + threadIdx.x;
  if (i >= B * TD * E) return;
  int row = i >> 8, e = i & (E - 1);
  int b = row / TD, t = row % TD;
  int tok = tgt[b * T + t];
  bf16_t v = (bf16_t)emb[(size_t)tok * E + e];
  dembs[i] = v;
  feats[(size_t)row * FK + 2 * H + e] = v;   // emb slice of logits features
}

__global__ void k_zero_tail(float* out) {
  int i = blockIdx.x * blockDim.x + threadIdx.x;
  if (i >= B * V) return;
  int b = i / V, j = i % V;
  out[((size_t)b * T + (T - 1)) * V + j] = 0.f;
}

// ---------------- generic WMMA GEMM:  C = A(MxK) * Bw(NxK)^T + bias ---------
// grid: (ceil(N/32), M/512), 256 threads. Each wave: 4 row tiles x 2 col tiles.
// M (grid.y*512) must be a multiple of 512; stores guarded by Mvalid.
__global__ __launch_bounds__(256) void k_gemm(
    const bf16_t* A, int lda, const bf16_t* Bw, int ldb, const float* bias,
    float* C, int ldc, bf16_t* Cbf, int ldcbf,
    int N, int K, int relu, int remap, int Mvalid) {
  const int wid  = threadIdx.x >> 5;
  const int l    = threadIdx.x & 31;
  const int n0   = blockIdx.x * 32;
  const int row0 = (blockIdx.y * 8 + wid) * 64;
  const bool has2 = (n0 + 16) < N;            // blockIdx-uniform scalar branch
  const int lm  = (l >> 4) * 8;
  const int nc0 = n0 + (l & 15);

  v8f acc[8] = {};                            // [0..3]=col0 rows, [4..7]=col1 rows
  if (has2) {
    // branch-free hot loop: 8 WMMAs per k-chunk, A fragments reused across 2 cols
    for (int k0 = 0; k0 < K; k0 += 32) {
      v16bf a0 = load_a_bf(A, lda, row0,      k0);
      v16bf a1 = load_a_bf(A, lda, row0 + 16, k0);
      v16bf a2 = load_a_bf(A, lda, row0 + 32, k0);
      v16bf a3 = load_a_bf(A, lda, row0 + 48, k0);
      v16bf b0 = load_b_bf(Bw, ldb, n0,      k0);
      v16bf b1 = load_b_bf(Bw, ldb, n0 + 16, k0);
      acc[0] = wmma_bf(a0, b0, acc[0]);
      acc[1] = wmma_bf(a1, b0, acc[1]);
      acc[2] = wmma_bf(a2, b0, acc[2]);
      acc[3] = wmma_bf(a3, b0, acc[3]);
      acc[4] = wmma_bf(a0, b1, acc[4]);
      acc[5] = wmma_bf(a1, b1, acc[5]);
      acc[6] = wmma_bf(a2, b1, acc[6]);
      acc[7] = wmma_bf(a3, b1, acc[7]);
    }
  } else {
    // ragged N edge (only the last block of the 30000-wide logits GEMM)
    for (int k0 = 0; k0 < K; k0 += 32) {
      v16bf b0 = load_b_bf(Bw, ldb, n0, k0);
      acc[0] = wmma_bf(load_a_bf(A, lda, row0,      k0), b0, acc[0]);
      acc[1] = wmma_bf(load_a_bf(A, lda, row0 + 16, k0), b0, acc[1]);
      acc[2] = wmma_bf(load_a_bf(A, lda, row0 + 32, k0), b0, acc[2]);
      acc[3] = wmma_bf(load_a_bf(A, lda, row0 + 48, k0), b0, acc[3]);
    }
  }

  for (int col = 0; col < 2; ++col) {
    if (col == 1 && !has2) break;
    int nc = nc0 + col * 16;
    float bv = bias ? bias[nc] : 0.f;
#pragma unroll
    for (int j = 0; j < 4; ++j) {
      v8f a = acc[col * 4 + j];
#pragma unroll
      for (int i = 0; i < 8; ++i) {
        int m = row0 + j * 16 + i + lm;
        if (m < Mvalid) {
          float v = a[i] + bv;
          if (relu) v = fmaxf(v, 0.f);
          size_t crow = remap ? ((size_t)(m / TD) * T + (m % TD)) : (size_t)m;
          C[crow * (size_t)ldc + nc] = v;
          if (Cbf) Cbf[(size_t)m * ldcbf + nc] = (bf16_t)v;
        }
      }
    }
  }
}

// ---------------- encoder recurrence (persistent, 8 WGs x 8 waves = 64 tiles)
__global__ __launch_bounds__(256) void k_enc_scan(char* ws, const int* lens,
                                                  const float* bhh0,
                                                  const float* bih1,
                                                  const float* bhh1) {
  float*  h0buf = (float*)(ws + OFF_H0);
  float*  h1buf = (float*)(ws + OFF_H1);
  bf16_t* h0bb  = (bf16_t*)(ws + OFF_H0B);
  bf16_t* h1bb  = (bf16_t*)(ws + OFF_H1B);
  const float*  gi0    = (const float*)(ws + OFF_GI0);
  const bf16_t* whh0   = (const bf16_t*)(ws + OFF_WHH0);
  const bf16_t* wih1   = (const bf16_t*)(ws + OFF_WIH1);
  const bf16_t* whh1   = (const bf16_t*)(ws + OFF_WHH1);
  bf16_t*       encraw = (bf16_t*)(ws + OFF_ENCRAW);
  int*          bar    = (int*)(ws + OFF_BAR);

  const int wid = threadIdx.x >> 5;
  const int l   = threadIdx.x & 31;
  const int p   = blockIdx.x * 8 + wid;          // 0..63 patch id
  const int row0 = (p >> 5) * 16;                // batch-row tile (0 or 16)
  const int n0   = (p & 31) * 16;                // hidden column tile
  const int lm = (l >> 4) * 8;
  const int nc = n0 + (l & 15);

  int rlen[8];
#pragma unroll
  for (int i = 0; i < 8; ++i) rlen[i] = lens[row0 + i + lm];

  const float bh0r = bhh0[nc], bh0z = bhh0[nc + H], bh0n = bhh0[nc + 2 * H];
  const float bi1r = bih1[nc], bi1z = bih1[nc + H], bi1n = bih1[nc + 2 * H];
  const float bh1r = bhh1[nc], bh1z = bhh1[nc + H], bh1n = bhh1[nc + 2 * H];

  for (int t = 0; t < S; ++t) {
    const float*  h0r  = h0buf + (t & 1) * (B * H);
    float*        h0w  = h0buf + ((t + 1) & 1) * (B * H);
    const bf16_t* h0rb = h0bb + (t & 1) * (B * H);
    bf16_t*       h0wb = h0bb + ((t + 1) & 1) * (B * H);
    const float*  h1r  = h1buf + (t & 1) * (B * H);
    float*        h1w  = h1buf + ((t + 1) & 1) * (B * H);
    const bf16_t* h1rb = h1bb + (t & 1) * (B * H);
    bf16_t*       h1wb = h1bb + ((t + 1) & 1) * (B * H);

    // layer 0: gh0 = h0 @ whh0^T  (3 gates, K=512)
    v8f ar = {}, az = {}, an = {};
    for (int k0 = 0; k0 < H; k0 += 32) {
      v16bf a = load_a_bf(h0rb, H, row0, k0);
      ar = wmma_bf(a, load_b_bf(whh0, H, n0,         k0), ar);
      az = wmma_bf(a, load_b_bf(whh0, H, n0 + H,     k0), az);
      an = wmma_bf(a, load_b_bf(whh0, H, n0 + 2 * H, k0), an);
    }
#pragma unroll
    for (int i = 0; i < 8; ++i) {
      int m = row0 + i + lm;
      size_t gb = ((size_t)m * S + t) * TH3 + nc;  // gi0 already includes bih0
      float hold = h0r[m * H + nc];
      float hnew = gru_gate(gi0[gb], gi0[gb + H], gi0[gb + 2 * H],
                            ar[i] + bh0r, az[i] + bh0z, an[i] + bh0n, hold);
      float hv = (t < rlen[i]) ? hnew : hold;
      h0w[m * H + nc]  = hv;
      h0wb[m * H + nc] = (bf16_t)hv;
    }
    gbarrier(bar + 0, bar + 1, NWG_ENC);

    // layer 1: gi1 = h0' @ wih1^T ; gh1 = h1 @ whh1^T
    v8f ir = {}, iz = {}, in_ = {}, hr = {}, hz = {}, hn = {};
    for (int k0 = 0; k0 < H; k0 += 32) {
      v16bf a0 = load_a_bf(h0wb, H, row0, k0);
      ir  = wmma_bf(a0, load_b_bf(wih1, H, n0,         k0), ir);
      iz  = wmma_bf(a0, load_b_bf(wih1, H, n0 + H,     k0), iz);
      in_ = wmma_bf(a0, load_b_bf(wih1, H, n0 + 2 * H, k0), in_);
      v16bf a1 = load_a_bf(h1rb, H, row0, k0);
      hr  = wmma_bf(a1, load_b_bf(whh1, H, n0,         k0), hr);
      hz  = wmma_bf(a1, load_b_bf(whh1, H, n0 + H,     k0), hz);
      hn  = wmma_bf(a1, load_b_bf(whh1, H, n0 + 2 * H, k0), hn);
    }
#pragma unroll
    for (int i = 0; i < 8; ++i) {
      int m = row0 + i + lm;
      float hold = h1r[m * H + nc];
      float hnew = gru_gate(ir[i] + bi1r, iz[i] + bi1z, in_[i] + bi1n,
                            hr[i] + bh1r, hz[i] + bh1z, hn[i] + bh1n, hold);
      bool v = (t < rlen[i]);
      float hv = v ? hnew : hold;
      h1w[m * H + nc]  = hv;
      h1wb[m * H + nc] = (bf16_t)hv;
      encraw[((size_t)m * S + t) * H + nc] = (bf16_t)(v ? hnew : 0.f);
    }
    gbarrier(bar + 0, bar + 1, NWG_ENC);
  }
}

// ---------------- decoder (persistent, 32 WGs x 8 waves) ---------------------
__global__ __launch_bounds__(256) void k_dec_scan(char* ws, const int* lens,
                                                  const float* attn_b,
                                                  const float* attn_vw,
                                                  const float* attn_vb,
                                                  const float* dec_bih,
                                                  const float* dec_bhh) {
  float*  dh    = (float*)(ws + OFF_DH);
  bf16_t* dhb   = (bf16_t*)(ws + OFF_DHB);
  float*  q     = (float*)(ws + OFF_Q);
  float*  score = (float*)(ws + OFF_SC);
  bf16_t* xcat  = (bf16_t*)(ws + OFF_XCAT);
  const bf16_t* dembs  = (const bf16_t*)(ws + OFF_DECEMB);
  const bf16_t* attnw  = (const bf16_t*)(ws + OFF_ATTNW);
  const bf16_t* dwih   = (const bf16_t*)(ws + OFF_DWIH);
  const bf16_t* dwhh   = (const bf16_t*)(ws + OFF_DWHH);
  const float*  encprj = (const float*)(ws + OFF_ENCPRJ);
  const float*  encout = (const float*)(ws + OFF_ENCOUT);
  const float*  h1f    = (const float*)(ws + OFF_H1);  // parity 0 = final enc h1
  bf16_t*       feats  = (bf16_t*)(ws + OFF_FEATS);
  int*          bar    = (int*)(ws + OFF_BAR);

  const int tid = threadIdx.x;
  const int wid = tid >> 5;
  const int l   = tid & 31;
  const int g   = blockIdx.x * 8 + wid;  // global wave id 0..255
  const int lm  = (l >> 4) * 8;

  __shared__ float sw[S];
  __shared__ float red[256];

  // init: prev_hidden = final encoder h1 (f32 + bf16 shadow, parity 0)
  {
    int b = blockIdx.x;
    for (int h = tid; h < H; h += 256) {
      float v = h1f[b * H + h];
      dh[b * H + h]  = v;
      dhb[b * H + h] = (bf16_t)v;
    }
  }
  gbarrier(bar + 2, bar + 3, NWG_DEC);

  const float vb = attn_vb[0];
  float vreg[16];
#pragma unroll
  for (int j = 0; j < 16; ++j) vreg[j] = attn_vw[l + 32 * j];

  for (int t = 0; t < TD; ++t) {
    // phase Q: q = emb_t @ attn_w[:,H:]^T + attn_b   (32x512, K=256)
    if (g < 64) {
      int row0 = (g >> 5) * 16, n0 = (g & 31) * 16;
      int nc = n0 + (l & 15);
      v8f acc = {};
      const bf16_t* Abase = dembs + (size_t)t * E;  // row stride TD*E
      for (int k0 = 0; k0 < E; k0 += 32) {
        v16bf a = load_a_bf(Abase, TD * E, row0, k0);
        v16bf b = load_b_bf(attnw + H, XK, n0, k0);
        acc = wmma_bf(a, b, acc);
      }
      float bq = attn_b[nc];
#pragma unroll
      for (int i = 0; i < 8; ++i) q[(row0 + i + lm) * H + nc] = acc[i] + bq;
    }
    gbarrier(bar + 2, bar + 3, NWG_DEC);

    // phase E: score[b,s] = v . tanh(enc_proj[b,s]+q[b]) + vb (masked)
    for (int pair = g; pair < B * S; pair += 256) {
      int b = pair >> 9, s2 = pair & (S - 1);
      float sc;
      if (s2 >= lens[b]) {
        sc = -__builtin_inff();
      } else {
        const float* pr = encprj + ((size_t)b * S + s2) * H;
        const float* qb = q + b * H;
        float sum = 0.f;
#pragma unroll
        for (int j = 0; j < 16; ++j) {
          int h = l + 32 * j;
          sum += vreg[j] * tanhf(pr[h] + qb[h]);
        }
        for (int o = 16; o; o >>= 1) sum += __shfl_xor(sum, o);
        sc = sum + vb;
      }
      if (l == 0) score[b * S + s2] = sc;
    }
    gbarrier(bar + 2, bar + 3, NWG_DEC);

    // phase S: softmax + context; build xcat = [emb|c], stash c into feats
    {
      int b = blockIdx.x;
      const float* sc = score + b * S;
      float m1 = fmaxf(sc[tid], sc[tid + 256]);
      red[tid] = m1; __syncthreads();
      for (int o = 128; o; o >>= 1) {
        if (tid < o) red[tid] = fmaxf(red[tid], red[tid + o]);
        __syncthreads();
      }
      float mx = red[0]; __syncthreads();
      float e0 = __expf(sc[tid] - mx), e1 = __expf(sc[tid + 256] - mx);
      sw[tid] = e0; sw[tid + 256] = e1;
      red[tid] = e0 + e1; __syncthreads();
      for (int o = 128; o; o >>= 1) {
        if (tid < o) red[tid] += red[tid + o];
        __syncthreads();
      }
      float inv = 1.f / red[0];
      sw[tid] *= inv; sw[tid + 256] *= inv;
      if (tid < E) xcat[b * XK + tid] = dembs[((size_t)b * TD + t) * E + tid];
      __syncthreads();
      for (int h = tid; h < H; h += 256) {
        float acc = 0.f;
        const float* eo = encout + (size_t)b * S * H + h;
        for (int s2 = 0; s2 < S; ++s2) acc += sw[s2] * eo[(size_t)s2 * H];
        bf16_t cb = (bf16_t)acc;
        xcat[b * XK + E + h] = cb;
        feats[((size_t)b * TD + t) * FK + H + h] = cb;
      }
    }
    gbarrier(bar + 2, bar + 3, NWG_DEC);

    // phase G: GRU cell (gi: K=768 over xcat, gh: K=512 over h)
    if (g < 64) {
      int row0 = (g >> 5) * 16, n0 = (g & 31) * 16;
      int nc = n0 + (l & 15);
      const float*  dhr  = dh  + (t & 1) * (B * H);
      float*        dhw  = dh  + ((t + 1) & 1) * (B * H);
      const bf16_t* dhrb = dhb + (t & 1) * (B * H);
      bf16_t*       dhwb = dhb + ((t + 1) & 1) * (B * H);
      v8f ir = {}, iz = {}, in_ = {};
      for (int k0 = 0; k0 < XK; k0 += 32) {
        v16bf a = load_a_bf(xcat, XK, row0, k0);
        ir  = wmma_bf(a, load_b_bf(dwih, XK, n0,         k0), ir);
        iz  = wmma_bf(a, load_b_bf(dwih, XK, n0 + H,     k0), iz);
        in_ = wmma_bf(a, load_b_bf(dwih, XK, n0 + 2 * H, k0), in_);
      }
      v8f hr = {}, hz = {}, hn = {};
      for (int k0 = 0; k0 < H; k0 += 32) {
        v16bf a = load_a_bf(dhrb, H, row0, k0);
        hr = wmma_bf(a, load_b_bf(dwhh, H, n0,         k0), hr);
        hz = wmma_bf(a, load_b_bf(dwhh, H, n0 + H,     k0), hz);
        hn = wmma_bf(a, load_b_bf(dwhh, H, n0 + 2 * H, k0), hn);
      }
      float bir = dec_bih[nc], biz = dec_bih[nc + H], bin = dec_bih[nc + 2 * H];
      float bhr = dec_bhh[nc], bhz = dec_bhh[nc + H], bhn = dec_bhh[nc + 2 * H];
#pragma unroll
      for (int i = 0; i < 8; ++i) {
        int m = row0 + i + lm;
        float hold = dhr[m * H + nc];
        float hnew = gru_gate(ir[i] + bir, iz[i] + biz, in_[i] + bin,
                              hr[i] + bhr, hz[i] + bhz, hn[i] + bhn, hold);
        dhw[m * H + nc]  = hnew;
        dhwb[m * H + nc] = (bf16_t)hnew;
        feats[((size_t)m * TD + t) * FK + nc] = (bf16_t)hnew;
      }
    }
    gbarrier(bar + 2, bar + 3, NWG_DEC);
  }
}

// ---------------- host orchestration ----------------------------------------
extern "C" void kernel_launch(void* const* d_in, const int* in_sizes, int n_in,
                              void* d_out, int out_size, void* d_ws, size_t ws_size,
                              hipStream_t stream) {
  (void)in_sizes; (void)n_in; (void)out_size; (void)ws_size;
  const int*   src    = (const int*)d_in[0];
  const int*   tgt    = (const int*)d_in[1];
  const int*   lens   = (const int*)d_in[2];
  const float* emb    = (const float*)d_in[3];
  const float* wih0   = (const float*)d_in[4];
  const float* whh0   = (const float*)d_in[5];
  const float* bih0   = (const float*)d_in[6];
  const float* bhh0   = (const float*)d_in[7];
  const float* wih1   = (const float*)d_in[8];
  const float* whh1   = (const float*)d_in[9];
  const float* bih1   = (const float*)d_in[10];
  const float* bhh1   = (const float*)d_in[11];
  const float* linw   = (const float*)d_in[12];
  const float* linb   = (const float*)d_in[13];
  const float* attnw  = (const float*)d_in[14];
  const float* attnb  = (const float*)d_in[15];
  const float* attnvw = (const float*)d_in[16];
  const float* attnvb = (const float*)d_in[17];
  const float* dwih   = (const float*)d_in[18];
  const float* dwhh   = (const float*)d_in[19];
  const float* dbih   = (const float*)d_in[20];
  const float* dbhh   = (const float*)d_in[21];
  const float* outw   = (const float*)d_in[22];
  const float* outb   = (const float*)d_in[23];
  char*  ws  = (char*)d_ws;
  float* out = (float*)d_out;

  {
    int ninit = (MFP - MF) * FK;  // largest init range (40960)
    if (ninit < B * H) ninit = B * H;
    k_init<<<(ninit + 255) / 256, 256, 0, stream>>>(ws);
  }

  struct CastJob { const float* s; size_t off; int n; };
  const CastJob jobs[9] = {
      {wih0, OFF_WIH0, TH3 * E}, {whh0, OFF_WHH0, TH3 * H},
      {wih1, OFF_WIH1, TH3 * H}, {whh1, OFF_WHH1, TH3 * H},
      {linw, OFF_LINW, H * H},   {attnw, OFF_ATTNW, H * XK},
      {dwih, OFF_DWIH, TH3 * XK},{dwhh, OFF_DWHH, TH3 * H},
      {outw, OFF_OUTW, V * FK}};
  for (int j = 0; j < 9; ++j)
    k_f2bf<<<(jobs[j].n + 255) / 256, 256, 0, stream>>>(
        jobs[j].s, (bf16_t*)(ws + jobs[j].off), jobs[j].n);

  k_gather_src<<<(B * S * E + 255) / 256, 256, 0, stream>>>(
      src, emb, (bf16_t*)(ws + OFF_SRCEMB));
  k_gather_tgt<<<(B * TD * E + 255) / 256, 256, 0, stream>>>(
      tgt, emb, (bf16_t*)(ws + OFF_DECEMB), (bf16_t*)(ws + OFF_FEATS));

  // gi0 = src_emb @ wih0^T + bih0  (all 512 steps at once)  M=16384
  k_gemm<<<dim3((TH3 + 31) / 32, (B * S) / 512), 256, 0, stream>>>(
      (const bf16_t*)(ws + OFF_SRCEMB), E, (const bf16_t*)(ws + OFF_WIH0), E,
      bih0, (float*)(ws + OFF_GI0), TH3, nullptr, 0, TH3, E, 0, 0, B * S);

  k_enc_scan<<<NWG_ENC, 256, 0, stream>>>(ws, lens, bhh0, bih1, bhh1);

  // enc_output = relu(enc_raw @ lin_w^T + lin_b)  (+ bf16 copy)
  k_gemm<<<dim3((H + 31) / 32, (B * S) / 512), 256, 0, stream>>>(
      (const bf16_t*)(ws + OFF_ENCRAW), H, (const bf16_t*)(ws + OFF_LINW), H,
      linb, (float*)(ws + OFF_ENCOUT), H, (bf16_t*)(ws + OFF_ENCOUTB), H,
      H, H, 1, 0, B * S);

  // enc_proj = enc_output @ attn_w[:, :H]^T
  k_gemm<<<dim3((H + 31) / 32, (B * S) / 512), 256, 0, stream>>>(
      (const bf16_t*)(ws + OFF_ENCOUTB), H, (const bf16_t*)(ws + OFF_ATTNW), XK,
      nullptr, (float*)(ws + OFF_ENCPRJ), H, nullptr, 0, H, H, 0, 0, B * S);

  k_dec_scan<<<NWG_DEC, 256, 0, stream>>>(ws, lens, attnb, attnvw, attnvb,
                                          dbih, dbhh);

  k_zero_tail<<<(B * V + 255) / 256, 256, 0, stream>>>(out);

  // logits: feats(2048x1280, 2016 valid) @ dec_out_w^T + b -> d_out (row remap)
  k_gemm<<<dim3((V + 31) / 32, MFP / 512), 256, 0, stream>>>(
      (const bf16_t*)(ws + OFF_FEATS), FK, (const bf16_t*)(ws + OFF_OUTW), FK,
      outb, out, V, nullptr, 0, V, FK, 0, 1, MF);
}
